// SpaceTransformer_nat_63814624084301
// MI455X (gfx1250) — compile-verified, matrix operand order checked by reading, not run on hardware
//
#include <hip/hip_runtime.h>
#include <cstdint>
#include <cstddef>

// ---------------------------------------------------------------------------
// Types for CDNA5 WMMA (wave32) + TDM descriptors
// ---------------------------------------------------------------------------
typedef __bf16 bf16_t;
typedef __attribute__((ext_vector_type(8)))  bf16_t v8bf;
typedef __attribute__((ext_vector_type(16))) bf16_t v16bf;
typedef __attribute__((ext_vector_type(8)))  float  v8f;
typedef __attribute__((ext_vector_type(4)))  unsigned int u32x4;
typedef __attribute__((ext_vector_type(8)))  int i32x8;
typedef __attribute__((ext_vector_type(4)))  int i32x4;

#define NTOK   18432      // 2*4*48*48 tokens
#define CDIM   128
#define HWPIX  2304       // 48*48
#define NHEAD  4

#if defined(__has_builtin)
#  if __has_builtin(__builtin_amdgcn_tensor_load_to_lds) && \
      __has_builtin(__builtin_amdgcn_s_wait_tensorcnt)
#    define HAVE_TDM 1
#  endif
#endif
#ifndef HAVE_TDM
#  define HAVE_TDM 0
#endif

// ---------------------------------------------------------------------------
// bf16 helpers (round-to-nearest-even), no reliance on bf16 arithmetic
// ---------------------------------------------------------------------------
__device__ __forceinline__ bf16_t f2bf(float f) {
  unsigned u = __builtin_bit_cast(unsigned, f);
  unsigned r = u + 0x7FFFu + ((u >> 16) & 1u);
  unsigned short s = (unsigned short)(r >> 16);
  return __builtin_bit_cast(bf16_t, s);
}
__device__ __forceinline__ float bf2f(bf16_t b) {
  unsigned short s = __builtin_bit_cast(unsigned short, b);
  unsigned u = ((unsigned)s) << 16;
  return __builtin_bit_cast(float, u);
}

__device__ __forceinline__ float wave_sum(float v) {
#pragma unroll
  for (int off = 16; off > 0; off >>= 1) v += __shfl_xor(v, off, 32);
  return v;
}
__device__ __forceinline__ float wave_max(float v) {
#pragma unroll
  for (int off = 16; off > 0; off >>= 1) v = fmaxf(v, __shfl_xor(v, off, 32));
  return v;
}

__device__ __forceinline__ float gelu_exact(float x) {
  return 0.5f * x * (1.0f + erff(x * 0.70710678118654752f));
}

// Per-lane WMMA fragment for a row-major bf16 matrix row:
// elements = [kk + 8*half .. +8) ++ [kk + 16 + 8*half .. +8)   (ISA 16-bit A layout)
__device__ __forceinline__ v16bf load_frag(const bf16_t* __restrict__ rowbase,
                                           int kk, int half) {
  v8bf c0 = *(const v8bf*)(rowbase + kk + 8 * half);
  v8bf c1 = *(const v8bf*)(rowbase + kk + 16 + 8 * half);
  v16bf r;
#pragma unroll
  for (int i = 0; i < 8; ++i) { r[i] = c0[i]; r[8 + i] = c1[i]; }
  return r;
}

// ---------------------------------------------------------------------------
// Weight f32 -> bf16 conversion (optionally scale the first `scale_n` elems)
// ---------------------------------------------------------------------------
__global__ void conv_bf16_kernel(const float* __restrict__ src,
                                 bf16_t* __restrict__ dst,
                                 int n, int scale_n, float scale) {
  int i = blockIdx.x * 256 + threadIdx.x;
  if (i < n) {
    float v = src[i];
    if (i < scale_n) v *= scale;
    dst[i] = f2bf(v);
  }
}

__global__ void scale_bias_kernel(const float* __restrict__ src,
                                  float* __restrict__ dst,
                                  int n, int scale_n, float scale) {
  int i = blockIdx.x * 256 + threadIdx.x;
  if (i < n) dst[i] = src[i] * (i < scale_n ? scale : 1.0f);
}

// ---------------------------------------------------------------------------
// LN1: gather-transpose x (B,C,D,H,W) -> t[N,C] f32, plus layernorm -> bf16
// One wave per token (32 lanes x 4 channels).
// ---------------------------------------------------------------------------
__global__ __launch_bounds__(256) void ln1_kernel(const float* __restrict__ x,
                                                  const float* __restrict__ wgt,
                                                  const float* __restrict__ bia,
                                                  float* __restrict__ t32,
                                                  bf16_t* __restrict__ ln) {
  const int n    = blockIdx.x * 8 + (threadIdx.x >> 5);
  const int lane = threadIdx.x & 31;
  const int bd = n / HWPIX, pix = n % HWPIX;
  const int b = bd >> 2, d = bd & 3;
  const size_t base = (size_t)b * (128 * 9216) + (size_t)d * HWPIX + pix;

  float v[4];
#pragma unroll
  for (int i = 0; i < 4; ++i) {
    const int c = lane + (i << 5);
    v[i] = x[base + (size_t)c * 9216];
  }
  float mean = wave_sum(v[0] + v[1] + v[2] + v[3]) * (1.0f / 128.0f);
  float vs = 0.0f;
#pragma unroll
  for (int i = 0; i < 4; ++i) { float dv = v[i] - mean; vs += dv * dv; }
  vs = wave_sum(vs);
  const float inv = rsqrtf(vs * (1.0f / 128.0f) + 1e-5f);
#pragma unroll
  for (int i = 0; i < 4; ++i) {
    const int c = lane + (i << 5);
    t32[(size_t)n * CDIM + c] = v[i];
    ln[(size_t)n * CDIM + c]  = f2bf((v[i] - mean) * inv * wgt[c] + bia[c]);
  }
}

// ---------------------------------------------------------------------------
// Residual + LN2: t2 = t + proj_out ; ln = LN(t2)
// ---------------------------------------------------------------------------
__global__ __launch_bounds__(256) void resid_ln2_kernel(const float* __restrict__ t,
                                                        const float* __restrict__ pr,
                                                        const float* __restrict__ wgt,
                                                        const float* __restrict__ bia,
                                                        float* __restrict__ t2,
                                                        bf16_t* __restrict__ ln) {
  const int n    = blockIdx.x * 8 + (threadIdx.x >> 5);
  const int lane = threadIdx.x & 31;
  float v[4];
#pragma unroll
  for (int i = 0; i < 4; ++i) {
    const int c = lane + (i << 5);
    v[i] = t[(size_t)n * CDIM + c] + pr[(size_t)n * CDIM + c];
  }
  float mean = wave_sum(v[0] + v[1] + v[2] + v[3]) * (1.0f / 128.0f);
  float vs = 0.0f;
#pragma unroll
  for (int i = 0; i < 4; ++i) { float dv = v[i] - mean; vs += dv * dv; }
  vs = wave_sum(vs);
  const float inv = rsqrtf(vs * (1.0f / 128.0f) + 1e-5f);
#pragma unroll
  for (int i = 0; i < 4; ++i) {
    const int c = lane + (i << 5);
    t2[(size_t)n * CDIM + c] = v[i];
    ln[(size_t)n * CDIM + c] = f2bf((v[i] - mean) * inv * wgt[c] + bia[c]);
  }
}

// ---------------------------------------------------------------------------
// bf16 WMMA GEMM: out[M,Nc] = A[M,Kd] @ W[Nc,Kd]^T + bias, fused epilogues.
// Block = 8 waves computing a 128x32 output tile (all waves share one 32-col
// weight tile). The weight tile (32*Kd bf16, contiguous rows) is staged into
// LDS once per block by the Tensor Data Mover; B fragments then come from LDS
// (ds_load_b128 broadcasts), A fragments stream from global (b128 + prefetch).
// mode: 0 = store bf16 ; 1 = store f32 ; 2 = GELU + store bf16
//       3 = add residual[M,128] and scatter f32 to (B,C,D,H,W) layout (Nc==128)
// ---------------------------------------------------------------------------
__global__ __launch_bounds__(256) void gemm_wmma(const bf16_t* __restrict__ A,
                                                 const bf16_t* __restrict__ W,
                                                 const float* __restrict__ bias,
                                                 const float* __restrict__ resid,
                                                 void* __restrict__ outp,
                                                 int M, int Kd, int Nc, int mode) {
  const int wid  = threadIdx.x >> 5;
  const int lane = threadIdx.x & 31;
  const int l16  = lane & 15, half = lane >> 4;
  const int blkpc = M >> 7;                      // blocks per column-tile strip
  const int nt = blockIdx.x / blkpc;
  const int mt = (blockIdx.x - nt * blkpc) * 8 + wid;
  const int cn = nt << 5;

  const bf16_t* arow = A + (size_t)(mt * 16 + l16) * Kd;

#if HAVE_TDM
  __shared__ bf16_t lds_w[32 * 256];             // 16 KB weight tile
  {
    // --- Tensor DMA descriptor: 1-D copy of 32*Kd bf16 elems (rows cn..cn+31)
    const unsigned elems = (unsigned)(32 * Kd);
    const unsigned lds_off = (unsigned)(uintptr_t)(&lds_w[0]);
    const unsigned long long ga =
        (unsigned long long)(uintptr_t)(W + (size_t)cn * (size_t)Kd);
    u32x4 g0;
    g0[0] = 1u;                                           // count=1 (valid D#)
    g0[1] = lds_off;                                      // lds_addr
    g0[2] = (unsigned)(ga & 0xFFFFFFFFull);               // global_addr lo
    g0[3] = (unsigned)((ga >> 32) & 0x01FFFFFFull) | (2u << 30);  // hi | type=2
    i32x8 g1;
    g1[0] = (int)(1u << 16);                              // data_size = 2 bytes
    g1[1] = (int)((elems & 0xFFFFu) << 16);               // tensor_dim0 lo16
    g1[2] = (int)(((elems >> 16) & 0xFFFFu) | (1u << 16)); // dim0 hi | dim1=1
    g1[3] = (int)((elems & 0xFFFFu) << 16);               // tile_dim0
    g1[4] = 1;                                            // tile_dim1 = 1
    g1[5] = (int)elems;                                   // dim0_stride lo
    g1[6] = 0; g1[7] = 0;
    i32x4 gz = {0, 0, 0, 0};
    if (threadIdx.x < 32) {                               // wave 0 issues DMA
#if __clang_major__ >= 23
      i32x8 g4 = {0, 0, 0, 0, 0, 0, 0, 0};
      __builtin_amdgcn_tensor_load_to_lds(g0, g1, gz, gz, g4, 0);
#else
      __builtin_amdgcn_tensor_load_to_lds(g0, g1, gz, gz, 0);
#endif
      __builtin_amdgcn_s_wait_tensorcnt(0);
    }
  }
  __syncthreads();
  const bf16_t* b0row = &lds_w[(size_t)l16 * Kd];
  const bf16_t* b1row = &lds_w[(size_t)(16 + l16) * Kd];
#else
  const bf16_t* b0row = W + (size_t)(cn + l16) * Kd;
  const bf16_t* b1row = W + (size_t)(cn + 16 + l16) * Kd;
#endif

  v8f acc0 = {}, acc1 = {};
  for (int kk = 0; kk < Kd; kk += 32) {
    __builtin_prefetch(arow + kk + 64, 0, 1);   // global_prefetch_b8 on A stream
    v16bf a  = load_frag(arow,  kk, half);
    v16bf b0 = load_frag(b0row, kk, half);
    v16bf b1 = load_frag(b1row, kk, half);
    acc0 = __builtin_amdgcn_wmma_f32_16x16x32_bf16(false, a, false, b0,
                                                   (short)0, acc0, false, false);
    acc1 = __builtin_amdgcn_wmma_f32_16x16x32_bf16(false, a, false, b1,
                                                   (short)0, acc1, false, false);
  }

  const float bia0 = bias[cn + l16];
  const float bia1 = bias[cn + 16 + l16];
  const int c0 = cn + l16, c1 = cn + 16 + l16;

#pragma unroll
  for (int r = 0; r < 8; ++r) {
    const int orow = mt * 16 + (half << 3) + r;   // C/D layout: row = 8*half + r
    float v0 = acc0[r] + bia0;
    float v1 = acc1[r] + bia1;
    if (mode == 2) { v0 = gelu_exact(v0); v1 = gelu_exact(v1); }
    if (mode == 0 || mode == 2) {
      bf16_t* o = (bf16_t*)outp;
      o[(size_t)orow * Nc + c0] = f2bf(v0);
      o[(size_t)orow * Nc + c1] = f2bf(v1);
    } else if (mode == 1) {
      float* o = (float*)outp;
      o[(size_t)orow * Nc + c0] = v0;
      o[(size_t)orow * Nc + c1] = v1;
    } else {
      // final: residual add + scatter to (B, C, D, H, W)
      float* o = (float*)outp;
      const int pix = orow % HWPIX, bd = orow / HWPIX;
      const int b = bd >> 2, dd = bd & 3;
      const size_t base = (size_t)b * (128 * 9216) + (size_t)dd * HWPIX + pix;
      o[base + (size_t)c0 * 9216] = v0 + resid[(size_t)orow * CDIM + c0];
      o[base + (size_t)c1 * 9216] = v1 + resid[(size_t)orow * CDIM + c1];
    }
  }
}

// ---------------------------------------------------------------------------
// Neighborhood attention: one wave per (token, head).
// qkv[N,384] bf16 (q pre-scaled by hd^-0.5). K=7 -> 49 neighbors, hd=32.
// ---------------------------------------------------------------------------
__global__ __launch_bounds__(256) void nat_attn_kernel(const bf16_t* __restrict__ qkv,
                                                       const float* __restrict__ rpb,
                                                       bf16_t* __restrict__ outp) {
  const int task = blockIdx.x * 8 + (threadIdx.x >> 5);
  const int lane = threadIdx.x & 31;
  const int n = task >> 2, hh = task & 3;
  const int bd = n / HWPIX, pix = n % HWPIX;
  const int h = pix / 48, w = pix % 48;
  int sh = h - 3; sh = sh < 0 ? 0 : (sh > 41 ? 41 : sh);
  int sw = w - 3; sw = sw < 0 ? 0 : (sw > 41 ? 41 : sw);

  const size_t qoff = (size_t)n * 384 + hh * 32;
  const float qd = bf2f(qkv[qoff + lane]);       // this lane's q dim

  // ---- scores for neighbors j0 = lane, j1 = lane + 32 -------------------
  const int j0 = lane, j1 = lane + 32;
  const int ih0 = sh + j0 / 7, iw0 = sw + j0 % 7;
  const int j1c = j1 < 49 ? j1 : j0;             // clamp for address safety
  const int ih1 = sh + j1c / 7, iw1 = sw + j1c % 7;
  const bf16_t* k0 = qkv + (size_t)(bd * HWPIX + ih0 * 48 + iw0) * 384 + 128 + hh * 32;
  const bf16_t* k1 = qkv + (size_t)(bd * HWPIX + ih1 * 48 + iw1) * 384 + 128 + hh * 32;

  float s0 = 0.0f, s1 = 0.0f;
#pragma unroll
  for (int d = 0; d < 32; ++d) {
    const float qv = __shfl(qd, d, 32);
    s0 += qv * bf2f(k0[d]);
    s1 += qv * bf2f(k1[d]);
  }
  s0 += rpb[hh * 169 + (ih0 - h + 6) * 13 + (iw0 - w + 6)];
  if (j1 < 49) s1 += rpb[hh * 169 + (ih1 - h + 6) * 13 + (iw1 - w + 6)];
  else         s1 = -1e30f;

  // ---- softmax over 49 values --------------------------------------------
  const float mx = wave_max(fmaxf(s0, s1));
  float p0 = __expf(s0 - mx);
  float p1 = (j1 < 49) ? __expf(s1 - mx) : 0.0f;
  const float inv = 1.0f / wave_sum(p0 + p1);
  p0 *= inv; p1 *= inv;

  // ---- AV: lane owns dim `lane` ------------------------------------------
  float o = 0.0f;
  const bf16_t* vbase = qkv + 256 + hh * 32 + lane;
#pragma unroll
  for (int j = 0; j < 49; ++j) {
    const int ih = sh + j / 7, iw = sw + j % 7;                 // scalar
    const size_t nn = (size_t)(bd * HWPIX + ih * 48 + iw) * 384;
    const float pj = (j < 32) ? __shfl(p0, j, 32) : __shfl(p1, j - 32, 32);
    o += pj * bf2f(vbase[nn]);
  }
  outp[(size_t)n * CDIM + hh * 32 + lane] = f2bf(o);
}

// ---------------------------------------------------------------------------
// Host-side orchestration
// ---------------------------------------------------------------------------
extern "C" void kernel_launch(void* const* d_in, const int* in_sizes, int n_in,
                              void* d_out, int out_size, void* d_ws, size_t ws_size,
                              hipStream_t stream) {
  const float* x     = (const float*)d_in[0];
  const float* n1w   = (const float*)d_in[1];
  const float* n1b   = (const float*)d_in[2];
  const float* qkvw  = (const float*)d_in[3];
  const float* qkvb  = (const float*)d_in[4];
  const float* rpb   = (const float*)d_in[5];
  const float* projw = (const float*)d_in[6];
  const float* projb = (const float*)d_in[7];
  const float* n2w   = (const float*)d_in[8];
  const float* n2b   = (const float*)d_in[9];
  const float* fc1w  = (const float*)d_in[10];
  const float* fc1b  = (const float*)d_in[11];
  const float* fc2w  = (const float*)d_in[12];
  const float* fc2b  = (const float*)d_in[13];

  char* ws = (char*)d_ws;
  float*  t32   = (float*)(ws + 0);                 // t  [N,128] f32  (reused as h1)
  float*  t2    = (float*)(ws + 9437184);           // t2 [N,128] f32
  bf16_t* lnb   = (bf16_t*)(ws + 18874368);         // ln [N,128] bf16 (ln1 then ln2)
  bf16_t* qkvo  = (bf16_t*)(ws + 23592960);         // qkv [N,384] bf16
  float*  projo = (float*)(ws + 23592960);          // proj out [N,128] f32 (reuses qkv)
  bf16_t* atto  = (bf16_t*)(ws + 37748736);         // attn out [N,128] bf16
  bf16_t* h1    = (bf16_t*)(ws + 0);                // mlp hidden [N,256] bf16 (reuses t)
  char*   wb    = ws + 42467328;
  bf16_t* qkvw_b  = (bf16_t*)(wb);                  // 384*128
  bf16_t* projw_b = (bf16_t*)(wb + 98304);          // 128*128
  bf16_t* fc1w_b  = (bf16_t*)(wb + 131072);         // 256*128
  bf16_t* fc2w_b  = (bf16_t*)(wb + 196608);         // 128*256
  float*  qkvb_s  = (float*)(wb + 262144);          // 384 f32

  const float qscale = 0.17677669529663687f;        // 32^-0.5 folded into q rows

  conv_bf16_kernel<<<192, 256, 0, stream>>>(qkvw, qkvw_b, 49152, 16384, qscale);
  conv_bf16_kernel<<<64,  256, 0, stream>>>(projw, projw_b, 16384, 0, 1.0f);
  conv_bf16_kernel<<<128, 256, 0, stream>>>(fc1w, fc1w_b, 32768, 0, 1.0f);
  conv_bf16_kernel<<<128, 256, 0, stream>>>(fc2w, fc2w_b, 32768, 0, 1.0f);
  scale_bias_kernel<<<2,  256, 0, stream>>>(qkvb, qkvb_s, 384, 128, qscale);

  // transpose + LN1
  ln1_kernel<<<NTOK / 8, 256, 0, stream>>>(x, n1w, n1b, t32, lnb);
  // QKV: [N,128] @ [128,384] -> bf16 (q pre-scaled)
  gemm_wmma<<<(NTOK / 128) * (384 / 32), 256, 0, stream>>>(
      lnb, qkvw_b, qkvb_s, nullptr, qkvo, NTOK, 128, 384, 0);
  // neighborhood attention
  nat_attn_kernel<<<NTOK * NHEAD / 8, 256, 0, stream>>>(qkvo, rpb, atto);
  // proj: [N,128] @ [128,128] -> f32
  gemm_wmma<<<(NTOK / 128) * (128 / 32), 256, 0, stream>>>(
      atto, projw_b, projb, nullptr, projo, NTOK, 128, 128, 1);
  // t2 = t + proj ; LN2
  resid_ln2_kernel<<<NTOK / 8, 256, 0, stream>>>(t32, projo, n2w, n2b, t2, lnb);
  // fc1 + GELU: [N,128] @ [128,256] -> bf16
  gemm_wmma<<<(NTOK / 128) * (256 / 32), 256, 0, stream>>>(
      lnb, fc1w_b, fc1b, nullptr, h1, NTOK, 128, 256, 2);
  // fc2 + residual + transpose back to (B,C,D,H,W) into d_out
  gemm_wmma<<<(NTOK / 128) * (128 / 32), 256, 0, stream>>>(
      h1, fc2w_b, fc2b, t2, d_out, NTOK, 256, 128, 3);
}
